// BilateralCorrelationFlex_68418829025448
// MI455X (gfx1250) — matrix operand order repaired and checked by reading, not compiled, not run on hardware
//
#include <hip/hip_runtime.h>

// ---------------------------------------------------------------------------
// BilateralCorrelationFlex for MI455X (gfx1250, wave32, WMMA bf16)
//
//   k_zero    : zero splat accumulators in ws
//   k_splat   : scatter-add barycentric-weighted prev_corr_feat (f32 atomics)
//   k_tables  : normalize + build bf16 gather tables splat1[2][4097][32],
//               splat2[2][4097][16]  (channel-contiguous rows)
//   k_weights : repack conv weights to bf16 WMMA-B row-major [out][K]; layer0
//               weights split into f-invariant (W0a, K=480 over splat1 chans)
//               and f-dependent (W0b, K=240->256 over splat2 chans) halves
//   k_main    : per 16-n tile (1 wave):
//               A1[16][480] <- splat1 gathers (once); base = A1 x W0a (30 WMMA)
//               for f in 0..14: A2[16][256] <- splat2 gathers;
//                 c = base + A2 x W0b (16 WMMA) -> leaky -> xs
//                 layer1 1x1 (2 WMMA) -> leaky -> xs
//                 blur0 accumulate (4 WMMA)
//               bias+leaky -> ys -> blur1 (8 WMMA) + bias -> d_out
//               => 368 WMMAs / tile, 512 tiles
// ---------------------------------------------------------------------------

typedef __attribute__((ext_vector_type(16))) __bf16 v16bf;
typedef __attribute__((ext_vector_type(8)))  __bf16 v8bf;
typedef __attribute__((ext_vector_type(8)))  float  v8f;

#define WMMA_BF16(a, b, c) \
    __builtin_amdgcn_wmma_f32_16x16x32_bf16(false, (a), false, (b), (short)0, (c), false, false)
#define SHUF16(lo, hi) \
    __builtin_shufflevector((lo), (hi), 0,1,2,3,4,5,6,7,8,9,10,11,12,13,14,15)

__device__ __forceinline__ float leaky(float v) { return v > 0.f ? v : v * 0.1f; }

// -------------------------------- k_zero -----------------------------------
__global__ void k_zero(float* __restrict__ p, int n) {
    int i = blockIdx.x * blockDim.x + threadIdx.x;
    if (i < n) p[i] = 0.f;
}

// -------------------------------- k_splat ----------------------------------
__global__ void k_splat(const float* __restrict__ bary,
                        const float* __restrict__ prev,
                        const int*   __restrict__ lat,
                        float* __restrict__ acc,   // [2*4097][16]
                        float* __restrict__ mass)  // [2*4097]
{
    int t = blockIdx.x * blockDim.x + threadIdx.x;      // over B*D1*Npts = 131072
    if (t >= 2 * 4 * 16384) return;
    int n  = t & 16383;
    int b  = t >> 16;                                    // t / (4*16384)
    float w   = bary[t];
    int   idx = lat[t] + 1;                              // flat in [0, 2*4097)
    unsafeAtomicAdd(&mass[idx], w);
    const float* pv = prev + (size_t)b * 16 * 16384 + n;
    float* a = acc + (size_t)idx * 16;
#pragma unroll
    for (int c = 0; c < 16; ++c)
        unsafeAtomicAdd(&a[c], w * pv[(size_t)c * 16384]);
}

// -------------------------------- k_tables ---------------------------------
__global__ void k_tables(const float* __restrict__ acc,
                         const float* __restrict__ mass,
                         const float* __restrict__ feat1,
                         const float* __restrict__ feat2,
                         __bf16* __restrict__ s1,     // [2*4097][32]
                         __bf16* __restrict__ s2)     // [2*4097][16]
{
    int r = blockIdx.x * blockDim.x + threadIdx.x;      // over 2*4097 = 8194 rows
    if (r >= 8194) return;
    int b = r / 4097, v = r % 4097;
    float inv = 1.f / (mass[r] + 1e-5f);
    __bf16* o1 = s1 + (size_t)r * 32;
    __bf16* o2 = s2 + (size_t)r * 16;
#pragma unroll
    for (int c = 0; c < 16; ++c)
        o1[c] = (__bf16)(acc[(size_t)r * 16 + c] * inv);
#pragma unroll
    for (int c = 0; c < 16; ++c) {
        float f1 = v ? feat1[((size_t)b * 16 + c) * 4096 + v - 1] : 0.f;
        float f2 = v ? feat2[((size_t)b * 16 + c) * 4096 + v - 1] : 0.f;
        o1[16 + c] = (__bf16)f1;
        o2[c]      = (__bf16)f2;
    }
}

// -------------------------------- k_weights --------------------------------
// W0a[32][480] : W0a[o][k*32+i]      = cw0[o][i][k]      (i < 32)
// W0b[32][256] : W0b[o][k*16+i]      = cw0[o][32+i][k]   (i < 16), pad 240..255
// W1 [32][32]  : cw1
// BW0[15][64][32] : BW0[f][o][c] = bw0[o][c][f]
// BW1[64][64]  : bw1
__global__ void k_weights(const float* __restrict__ cw0, const float* __restrict__ cw1,
                          const float* __restrict__ bw0, const float* __restrict__ bw1,
                          __bf16* __restrict__ W0a, __bf16* __restrict__ W0b,
                          __bf16* __restrict__ W1,
                          __bf16* __restrict__ BW0, __bf16* __restrict__ BW1)
{
    int t = blockIdx.x * blockDim.x + threadIdx.x;      // 59392 total
    if (t < 15360) {
        int o = t / 480, kk = t % 480;
        int k = kk / 32, i = kk % 32;
        W0a[t] = (__bf16)cw0[((size_t)o * 48 + i) * 15 + k];
    } else if (t < 23552) {
        int u = t - 15360;
        int o = u / 256, kk = u % 256;
        float v = 0.f;
        if (kk < 240) { int k = kk / 16, i = kk % 16; v = cw0[((size_t)o * 48 + 32 + i) * 15 + k]; }
        W0b[u] = (__bf16)v;
    } else if (t < 24576) {
        W1[t - 23552] = (__bf16)cw1[t - 23552];
    } else if (t < 55296) {
        int u = t - 24576;
        int f = u / 2048, rem = u % 2048;
        int o = rem / 32, c = rem % 32;
        BW0[u] = (__bf16)bw0[((size_t)o * 32 + c) * 15 + f];
    } else if (t < 59392) {
        BW1[t - 55296] = (__bf16)bw1[t - 55296];
    }
}

// -------------------------------- k_main -----------------------------------
__global__ __launch_bounds__(128) void k_main(
    const __bf16* __restrict__ s1, const __bf16* __restrict__ s2,
    const __bf16* __restrict__ W0a, const __bf16* __restrict__ W0b,
    const __bf16* __restrict__ W1,
    const __bf16* __restrict__ BW0, const __bf16* __restrict__ BW1,
    const int* __restrict__ pc1, const int* __restrict__ pc2,
    const float* __restrict__ cb0, const float* __restrict__ cb1,
    const float* __restrict__ bb0, const float* __restrict__ bb1,
    float* __restrict__ out)
{
    extern __shared__ __align__(32) char smem_raw[];
    __bf16* smem = (__bf16*)smem_raw;

    const int lane = threadIdx.x & 31;
    const int wave = threadIdx.x >> 5;
    const int tile = blockIdx.x * 4 + wave;     // 512 tiles = 2 batches * 256
    const int b    = tile >> 8;
    const int nb   = (tile & 255) << 4;         // n base (16 rows per tile)

    __bf16* A1 = smem + wave * 13312;           // [16][480] bf16 (splat1 part)
    __bf16* A2 = A1 + 7680;                     // [16][256] bf16 (splat2 part)
    __bf16* xs = A1 + 11776;                    // [16][32]  bf16 staging
    __bf16* ys = A1 + 12288;                    // [16][64]  bf16 staging

    const int half = lane >> 4;                 // 0 | 1
    const int ln   = lane & 15;
    const int kbA  = half * 16;                 // A-frag K sub-base
    const int kbB  = half * 16;                 // B-frag K sub-base

    // per-lane biases
    const float cb0_0 = cb0[ln], cb0_1 = cb0[16 + ln];
    const float cb1_0 = cb1[ln], cb1_1 = cb1[16 + ln];
    float bb0v[4], bb1v[4];
#pragma unroll
    for (int ot = 0; ot < 4; ++ot) { bb0v[ot] = bb0[ot * 16 + ln]; bb1v[ot] = bb1[ot * 16 + ln]; }

    // ---- stage splat1 gathers (f-invariant): A1[r][k*32 .. +31] ----
    for (int t = lane; t < 240; t += 32) {
        int k = t >> 4, r = t & 15;
        int j = pc1[((size_t)b * 15 + k) * 4096 + nb + r] + 1;     // [0,4096]
        const int4* src = (const int4*)(s1 + ((size_t)b * 4097 + j) * 32);
        int4* dst = (int4*)(A1 + r * 480 + k * 32);
        dst[0] = src[0]; dst[1] = src[1]; dst[2] = src[2]; dst[3] = src[3];
    }
    if (lane < 16) {                            // zero A2 K pad 240..255 (once)
        int4 z = {0, 0, 0, 0};
        int4* dst = (int4*)(A2 + lane * 256 + 240);
        dst[0] = z; dst[1] = z;
    }
    __builtin_amdgcn_wave_barrier();
    asm volatile("s_wait_dscnt 0" ::: "memory");

    // ---- f-invariant base partial product: [16x480] x [480x32] ----
    v8f base0 = {}, base1 = {};
    for (int ks = 0; ks < 15; ++ks) {
        int kk = ks * 32;
        v8bf alo = *(const v8bf*)(A1 + ln * 480 + kk + kbA);
        v8bf ahi = *(const v8bf*)(A1 + ln * 480 + kk + kbA + 8);
        v16bf a  = SHUF16(alo, ahi);
        v16bf w0 = *(const v16bf*)(W0a + (size_t)ln * 480 + kk + kbB);
        v16bf w1 = *(const v16bf*)(W0a + (size_t)(16 + ln) * 480 + kk + kbB);
        base0 = WMMA_BF16(a, w0, base0);
        base1 = WMMA_BF16(a, w1, base1);
    }

    v8f yacc[4] = {};
    const int* pc2b = pc2 + (size_t)b * 15 * 15 * 4096 + nb;

    for (int f = 0; f < 15; ++f) {
        // ---- stage splat2 gathers: A2[r][k*16 .. +15] ----
        for (int t = lane; t < 240; t += 32) {
            int k = t >> 4, r = t & 15;
            int j = pc2b[((size_t)f * 15 + k) * 4096 + r] + 1;
            const int4* src = (const int4*)(s2 + ((size_t)b * 4097 + j) * 16);
            int4* dst = (int4*)(A2 + r * 256 + k * 16);
            dst[0] = src[0]; dst[1] = src[1];
        }
        __builtin_amdgcn_wave_barrier();
        asm volatile("s_wait_dscnt 0" ::: "memory");

        // ---- corr layer0 (f-dependent half): c = base + [16x256]x[256x32] ----
        v8f c0 = base0, c1 = base1;
        for (int ks = 0; ks < 8; ++ks) {
            int kk = ks * 32;
            v8bf alo = *(const v8bf*)(A2 + ln * 256 + kk + kbA);
            v8bf ahi = *(const v8bf*)(A2 + ln * 256 + kk + kbA + 8);
            v16bf a  = SHUF16(alo, ahi);
            v16bf w0 = *(const v16bf*)(W0b + (size_t)ln * 256 + kk + kbB);
            v16bf w1 = *(const v16bf*)(W0b + (size_t)(16 + ln) * 256 + kk + kbB);
            c0 = WMMA_BF16(a, w0, c0);
            c1 = WMMA_BF16(a, w1, c1);
        }
#pragma unroll
        for (int r = 0; r < 8; ++r) {                // bias + leaky -> xs[16][32]
            int m = r + half * 8;
            xs[m * 32 + ln]      = (__bf16)leaky(c0[r] + cb0_0);
            xs[m * 32 + 16 + ln] = (__bf16)leaky(c1[r] + cb0_1);
        }
        __builtin_amdgcn_wave_barrier();
        asm volatile("s_wait_dscnt 0" ::: "memory");

        // ---- corr layer1 (1x1): [16x32] x [32x32] ----
        v16bf a1;
        {
            v8bf alo = *(const v8bf*)(xs + ln * 32 + kbA);
            v8bf ahi = *(const v8bf*)(xs + ln * 32 + kbA + 8);
            a1 = SHUF16(alo, ahi);
        }
        v16bf w10 = *(const v16bf*)(W1 + ln * 32 + kbB);
        v16bf w11 = *(const v16bf*)(W1 + (16 + ln) * 32 + kbB);
        v8f d0 = {}, d1 = {};
        d0 = WMMA_BF16(a1, w10, d0);
        d1 = WMMA_BF16(a1, w11, d1);
#pragma unroll
        for (int r = 0; r < 8; ++r) {                // bias + leaky -> xs[16][32]
            int m = r + half * 8;
            xs[m * 32 + ln]      = (__bf16)leaky(d0[r] + cb1_0);
            xs[m * 32 + 16 + ln] = (__bf16)leaky(d1[r] + cb1_1);
        }
        __builtin_amdgcn_wave_barrier();
        asm volatile("s_wait_dscnt 0" ::: "memory");

        // ---- blur layer0 accumulate over f: [16x32] x [32x64] ----
        v16bf a2;
        {
            v8bf alo = *(const v8bf*)(xs + ln * 32 + kbA);
            v8bf ahi = *(const v8bf*)(xs + ln * 32 + kbA + 8);
            a2 = SHUF16(alo, ahi);
        }
        const __bf16* bwf = BW0 + (size_t)f * 2048;  // [64][32]
#pragma unroll
        for (int ot = 0; ot < 4; ++ot) {
            v16bf wv = *(const v16bf*)(bwf + (ot * 16 + ln) * 32 + kbB);
            yacc[ot] = WMMA_BF16(a2, wv, yacc[ot]);
        }
    }

    // ---- blur0 bias + leaky -> ys[16][64] ----
#pragma unroll
    for (int r = 0; r < 8; ++r) {
        int m = r + half * 8;
        ys[m * 64 + ln]      = (__bf16)leaky(yacc[0][r] + bb0v[0]);
        ys[m * 64 + 16 + ln] = (__bf16)leaky(yacc[1][r] + bb0v[1]);
        ys[m * 64 + 32 + ln] = (__bf16)leaky(yacc[2][r] + bb0v[2]);
        ys[m * 64 + 48 + ln] = (__bf16)leaky(yacc[3][r] + bb0v[3]);
    }
    __builtin_amdgcn_wave_barrier();
    asm volatile("s_wait_dscnt 0" ::: "memory");

    // ---- blur layer1 (no relu): [16x64] x [64x64] ----
    v8f z[4] = {};
#pragma unroll
    for (int ks = 0; ks < 2; ++ks) {
        v8bf alo = *(const v8bf*)(ys + ln * 64 + ks * 32 + kbA);
        v8bf ahi = *(const v8bf*)(ys + ln * 64 + ks * 32 + kbA + 8);
        v16bf a  = SHUF16(alo, ahi);
#pragma unroll
        for (int ot = 0; ot < 4; ++ot) {
            v16bf wv = *(const v16bf*)(BW1 + (ot * 16 + ln) * 64 + ks * 32 + kbB);
            z[ot] = WMMA_BF16(a, wv, z[ot]);
        }
    }

    // ---- store: out[b][o][n], lane holds o = ot*16+ln, n = nb + half*8 + r ----
#pragma unroll
    for (int ot = 0; ot < 4; ++ot) {
        int o = ot * 16 + ln;
        float bias = bb1v[ot];
        float* op = out + ((size_t)b * 64 + o) * 4096 + nb + half * 8;
        float4 v0 = make_float4(z[ot][0] + bias, z[ot][1] + bias, z[ot][2] + bias, z[ot][3] + bias);
        float4 v1 = make_float4(z[ot][4] + bias, z[ot][5] + bias, z[ot][6] + bias, z[ot][7] + bias);
        *(float4*)op       = v0;
        *((float4*)op + 1) = v1;
    }
}

// ------------------------------ kernel_launch ------------------------------
extern "C" void kernel_launch(void* const* d_in, const int* in_sizes, int n_in,
                              void* d_out, int out_size, void* d_ws, size_t ws_size,
                              hipStream_t stream)
{
    const float* feat1 = (const float*)d_in[0];
    const float* feat2 = (const float*)d_in[1];
    const float* prev  = (const float*)d_in[2];
    const float* bary  = (const float*)d_in[3];
    const float* cw0   = (const float*)d_in[4];
    const float* cb0   = (const float*)d_in[5];
    const float* cw1   = (const float*)d_in[6];
    const float* cb1   = (const float*)d_in[7];
    const float* bw0   = (const float*)d_in[8];
    const float* bb0   = (const float*)d_in[9];
    const float* bw1   = (const float*)d_in[10];
    const float* bb1   = (const float*)d_in[11];
    const int*   lat   = (const int*)d_in[12];
    const int*   pc1   = (const int*)d_in[13];
    const int*   pc2   = (const int*)d_in[14];

    // workspace layout
    float*  acc  = (float*)d_ws;                       // 131104 f32 (8194 x 16)
    float*  mass = acc + 131104;                       //   8194 f32
    __bf16* bf   = (__bf16*)((char*)d_ws + 557312);    // 32B-aligned bf16 region
    __bf16* S1   = bf;                                 // 8194*32 = 262208
    __bf16* S2   = bf + 262208;                        // 8194*16 = 131104
    __bf16* W0a  = bf + 393312;                        // 32*480  =  15360
    __bf16* W0b  = bf + 408672;                        // 32*256  =   8192
    __bf16* W1   = bf + 416864;                        // 32*32   =   1024
    __bf16* BW0  = bf + 417888;                        // 15*64*32=  30720
    __bf16* BW1  = bf + 448608;                        // 64*64   =   4096

    k_zero   <<<545, 256, 0, stream>>>(acc, 139298);
    k_splat  <<<512, 256, 0, stream>>>(bary, prev, lat, acc, mass);
    k_tables <<<33, 256, 0, stream>>>(acc, mass, feat1, feat2, S1, S2);
    k_weights<<<232, 256, 0, stream>>>(cw0, cw1, bw0, bw1, W0a, W0b, W1, BW0, BW1);
    k_main   <<<128, 128, 106496, stream>>>(S1, S2, W0a, W0b, W1, BW0, BW1,
                                            pc1, pc2, cb0, cb1, bb0, bb1,
                                            (float*)d_out);
}
